// Attention_75136157876852
// MI455X (gfx1250) — compile-verified, matrix operand order checked by reading, not run on hardware
//
#include <hip/hip_runtime.h>

typedef __attribute__((ext_vector_type(16))) __bf16   v16bf;
typedef __attribute__((ext_vector_type(8)))  float    v8f;
typedef __attribute__((ext_vector_type(4)))  unsigned uv4;
typedef __attribute__((ext_vector_type(8)))  int      iv8;
typedef __attribute__((ext_vector_type(4)))  int      iv4;
typedef __attribute__((ext_vector_type(4)))  float    fv4;
typedef unsigned short u16;

#define BATCH 8
#define SEQ   4096
#define CH    512
#define NHEAD 8
#define LH    512            // SEQ / NHEAD
#define NQKV  1536
#define ATT_SCALE 0.125f     // 64^-0.5
#define HEADSZ ((size_t)CH * LH)   // elements per (b,h) panel = 262144

// workspace layout, in u16 elements
#define WS_WQ  0                                   // w_qkv bf16 [1536][512]
#define WS_WO  (NQKV * CH)                         // w_out bf16 [512][512]
#define WS_QT  (WS_WO + CH * CH)                   // qT [b*h][c][m]  (pre-scaled)
#define WS_KT  (WS_QT + BATCH * NHEAD * CH * LH)   // kT [b*h][c][m]
#define WS_VS  (WS_KT + BATCH * NHEAD * CH * LH)   // vS [b*h][m][c]
#define WS_YR  (WS_VS + BATCH * NHEAD * LH * CH)   // yR [b][m*h+hh][c]

__device__ __forceinline__ u16 f2bf(float x) {
  unsigned u = __builtin_bit_cast(unsigned, x);
  return (u16)((u + 0x7FFFu + ((u >> 16) & 1u)) >> 16);
}

union Frag {
  v16bf v;
  uv4   q[2];
  u16   s[16];
};

// K-major fragment load for CDNA5 16-bit WMMA operands.
// Lane L: row = L&15; elems 0..7 = K[(L>>4)*8 .. +7]; elems 8..15 = K[16+(L>>4)*8 ..].
__device__ __forceinline__ Frag load_frag(const u16* p, int row0, int ld, int k0, int lane) {
  int r  = lane & 15;
  int hi = lane >> 4;
  const u16* q = p + (size_t)(row0 + r) * ld + k0 + hi * 8;
  Frag f;
  f.q[0] = *(const uv4*)(q);
  f.q[1] = *(const uv4*)(q + 16);
  return f;
}

__device__ __forceinline__ v8f wmma_bf16(const Frag& a, const Frag& b, v8f c) {
  return __builtin_amdgcn_wmma_f32_16x16x32_bf16(false, a.v, false, b.v,
                                                 (short)0, c, false, false);
}

// Tensor Data Mover: 2-D bf16 tile load (tile_rows x tile_cols) global -> LDS,
// row-major compaction in LDS. D# packing per CDNA5 ISA ch.8.
// This toolchain exposes the 6-arg builtin: (g0, g1, g2, g3, g4, cpol).
__device__ __forceinline__ void tdm_load_2d(const u16* gsrc, unsigned lds_byte,
                                            int cols, int rows, int ld_elems) {
  unsigned long long ga = (unsigned long long)(const void*)gsrc;
  uv4 g0;
  g0[0] = 1u;                                        // count=1, is_restore=0
  g0[1] = lds_byte;                                  // lds_addr
  g0[2] = (unsigned)(ga & 0xFFFFFFFFu);              // global_addr[31:0]
  g0[3] = (unsigned)((ga >> 32) & 0x1FFFFFFu)        // global_addr[56:32]
          | (2u << 30);                              // type=2 (image)
  iv8 g1;
  g1[0] = (int)(1u << 16);                           // wg_mask=0, data_size=1 (2B)
  g1[1] = (int)(((unsigned)cols & 0xFFFFu) << 16);   // tensor_dim0[15:0] in [31:16]
  g1[2] = (int)((((unsigned)cols >> 16) & 0xFFFFu)   // tensor_dim0[31:16]
          | (((unsigned)rows & 0xFFFFu) << 16));     // tensor_dim1[15:0]
  g1[3] = (int)((((unsigned)rows >> 16) & 0xFFFFu)   // tensor_dim1[31:16]
          | (((unsigned)cols & 0xFFFFu) << 16));     // tile_dim0
  g1[4] = (int)((unsigned)rows & 0xFFFFu);           // tile_dim1, tile_dim2=0
  g1[5] = (int)(unsigned)ld_elems;                   // tensor_dim0_stride[31:0]
  g1[6] = 0;                                         // stride hi, dim1_stride lo
  g1[7] = 0;
  iv4 z4 = (iv4)0;
  iv8 z8 = (iv8)0;
  __builtin_amdgcn_tensor_load_to_lds(g0, g1, z4, z4, z8, 0);
}

// ---------------------------------------------------------------------------
// Kernel 0: convert both weight matrices f32 -> bf16 into ws
// ---------------------------------------------------------------------------
__global__ __launch_bounds__(256) void cvt_weights(const float* __restrict__ wqkv,
                                                   const float* __restrict__ wout,
                                                   u16* __restrict__ ws) {
  int id = blockIdx.x * 256 + threadIdx.x;
  const int NW = NQKV * CH;
  if (id < NW) {
    ws[WS_WQ + id] = f2bf(wqkv[id]);
  } else {
    int j = id - NW;
    ws[WS_WO + j] = f2bf(wout[j]);
  }
}

// ---------------------------------------------------------------------------
// Kernel 1: qkv = x @ w_qkv^T  (M=32768, N=1536, K=512), scatter to q/k/v
// 128x128 tile per workgroup, 8 waves (4M x 2N), 32x64 per wave.
// ---------------------------------------------------------------------------
__global__ __launch_bounds__(256, 2) void qkv_gemm(const float* __restrict__ x,
                                                   const u16* __restrict__ wq,
                                                   u16* __restrict__ qT,
                                                   u16* __restrict__ kT,
                                                   u16* __restrict__ vS) {
  __shared__ __align__(16) u16 As[128 * 32];
  __shared__ __align__(16) u16 Bs[128 * 32];
  const int tid = threadIdx.x, lane = tid & 31, wave = tid >> 5;
  const int wm = wave & 3, wn = wave >> 2;
  const int row0g = blockIdx.x * 128;
  const int col0g = blockIdx.y * 128;

  v8f acc[2][4] = {};
  for (int kk = 0; kk < CH; kk += 32) {
    // stage A: 128x32 of x, f32 -> bf16 (4 independent b128 loads per thread)
    {
      fv4 f[4];
#pragma unroll
      for (int it = 0; it < 4; ++it) {
        int i = tid + it * 256;
        int r = i >> 3, c = i & 7;
        f[it] = *(const fv4*)&x[(size_t)(row0g + r) * CH + kk + c * 4];
      }
#pragma unroll
      for (int it = 0; it < 4; ++it) {
        int i = tid + it * 256;
        int r = i >> 3, c = i & 7;
        u16* d = &As[r * 32 + c * 4];
        d[0] = f2bf(f[it].x); d[1] = f2bf(f[it].y);
        d[2] = f2bf(f[it].z); d[3] = f2bf(f[it].w);
      }
    }
    // stage B: 128 output-cols x 32 K of w_qkv (2 independent b128 copies)
    {
      uv4 t[2];
#pragma unroll
      for (int it = 0; it < 2; ++it) {
        int i = tid + it * 256;
        int r = i >> 2, c = i & 3;
        t[it] = *(const uv4*)&wq[(size_t)(col0g + r) * CH + kk + c * 8];
      }
#pragma unroll
      for (int it = 0; it < 2; ++it) {
        int i = tid + it * 256;
        int r = i >> 2, c = i & 3;
        *(uv4*)&Bs[r * 32 + c * 8] = t[it];
      }
    }
    __syncthreads();
    Frag a0 = load_frag(As, wm * 32, 32, 0, lane);
    Frag a1 = load_frag(As, wm * 32 + 16, 32, 0, lane);
    for (int ni = 0; ni < 4; ++ni) {            // one B fragment live at a time
      Frag b = load_frag(Bs, wn * 64 + ni * 16, 32, 0, lane);
      acc[0][ni] = wmma_bf16(a0, b, acc[0][ni]);
      acc[1][ni] = wmma_bf16(a1, b, acc[1][ni]);
    }
    __syncthreads();
  }

  // Scatter epilogue. Within a C-tile register column, i varies only the head
  // (hh == i, bases are multiples of 8), so every destination is
  // base + i*HEADSZ for q, k and v alike -> constant-stride stores, no spills.
  const int s = col0g >> 9;          // 0=q, 1=k, 2=v (uniform per block)
  const int cb = col0g & 511;
  const int hi = lane >> 4, ln = lane & 15;
  const float mult = (s == 0) ? ATT_SCALE : 1.f;
  u16* dst = (s == 0) ? qT : (s == 1) ? kT : vS;
  for (int mi = 0; mi < 2; ++mi) {
    int gr0 = row0g + wm * 32 + mi * 16 + 8 * hi;   // multiple of 8
    int bb  = gr0 >> 12;
    int m   = (gr0 & 4095) >> 3;
    for (int ni = 0; ni < 4; ++ni) {
      int cc = cb + wn * 64 + ni * 16 + ln;
      size_t base = (size_t)bb * (NHEAD * HEADSZ)
                  + (s == 2 ? (size_t)m * CH + cc : (size_t)cc * LH + m);
#pragma unroll
      for (int i = 0; i < 8; ++i)
        dst[base + (size_t)i * HEADSZ] = f2bf(acc[mi][ni][i] * mult);
    }
  }
}

// ---------------------------------------------------------------------------
// Kernel 2: fused  S = q·k^T -> softmax -> y = P·v   per (b,h, 16-row stripe)
// grid = (32 stripes, 64 heads), 8 waves; wave w owns columns [64w, 64w+64).
// All WMMA operands are TDM-fed from LDS:
//   R1 (32KB): K panel (GEMM1) -> Sf scores (softmax) -> V panel (GEMM2)
//   R2 (16KB): Q stripe (GEMM1) -> P probabilities (GEMM2 A operand)
// ---------------------------------------------------------------------------
__global__ __launch_bounds__(256, 2) void attn_fused(const u16* __restrict__ qT,
                                                     const u16* __restrict__ kT,
                                                     const u16* __restrict__ vS,
                                                     u16* __restrict__ yR) {
  __shared__ __align__(16) unsigned char R1[512 * 32 * 2];  // 32 KB panel region
  __shared__ __align__(16) u16 R2[16 * 512];                // 16 KB Q/P region
  u16*   Panel = (u16*)R1;
  float* Sf    = (float*)R1;
  const int tid = threadIdx.x, lane = tid & 31, w = tid >> 5;
  const int hi = lane >> 4, ln = lane & 15;
  const int ci0 = blockIdx.x * 16;
  const int bh  = blockIdx.y;
  const u16* qh = qT + (size_t)bh * HEADSZ;
  const u16* kh = kT + (size_t)bh * HEADSZ;
  const u16* vh = vS + (size_t)bh * HEADSZ;
  const unsigned lds_panel = (unsigned)(unsigned long long)(const void*)Panel;
  const unsigned lds_q     = (unsigned)(unsigned long long)(const void*)R2;

  // TDM: DMA the 16x512 bf16 q stripe into LDS (one descriptor, wave 0 only)
  if (w == 0) {
    tdm_load_2d(qh + (size_t)ci0 * LH, lds_q, LH, 16, LH);
    __builtin_amdgcn_s_wait_tensorcnt(0);
  }
  __syncthreads();

  // S[ci, d] = sum_m q[ci,m] * k[d,m]   (scale pre-folded into q)
  v8f accS[4] = {};
  for (int kk = 0; kk < LH; kk += 32) {
    if (w == 0) {           // DMA the 512x32 k panel for this K step
      tdm_load_2d(kh + kk, lds_panel, 32, CH, LH);
      __builtin_amdgcn_s_wait_tensorcnt(0);
    }
    __syncthreads();        // panel ready
    Frag a = load_frag(R2, 0, LH, kk, lane);
    for (int ni = 0; ni < 4; ++ni) {
      Frag b = load_frag(Panel, w * 64 + ni * 16, 32, 0, lane);
      accS[ni] = wmma_bf16(a, b, accS[ni]);
    }
    __syncthreads();        // all panel reads done before next DMA overwrites
  }
  for (int ni = 0; ni < 4; ++ni)
    for (int i = 0; i < 8; ++i)
      Sf[(i + 8 * hi) * 512 + w * 64 + ni * 16 + ln] = accS[ni][i];
  __syncthreads();          // Q stripe dead; R2 becomes P

  // row-wise softmax over 512 cols; 16 threads per row (lane groups of 16)
  {
    int row = tid >> 4, tc = tid & 15;
    float mx = -3.0e38f;
    for (int j = tc; j < 512; j += 16) mx = fmaxf(mx, Sf[row * 512 + j]);
    for (int d = 8; d; d >>= 1) mx = fmaxf(mx, __shfl_xor(mx, d, 16));
    float sum = 0.f;
    for (int j = tc; j < 512; j += 16) {
      float e = __expf(Sf[row * 512 + j] - mx);
      Sf[row * 512 + j] = e;
      sum += e;
    }
    for (int d = 8; d; d >>= 1) sum += __shfl_xor(sum, d, 16);
    float inv = 1.f / sum;
    for (int j = tc; j < 512; j += 16) R2[row * 512 + j] = f2bf(Sf[row * 512 + j] * inv);
  }
  __syncthreads();          // Sf dead; R1 becomes the V panel

  // y[ci, m] = sum_d P[ci,d] * v[d,m]   (vS is [m][d]: row-major panels)
  v8f accY[4] = {};
  for (int kk = 0; kk < CH; kk += 32) {
    if (w == 0) {           // DMA the 512x32 v panel for this K step
      tdm_load_2d(vh + kk, lds_panel, 32, LH, CH);
      __builtin_amdgcn_s_wait_tensorcnt(0);
    }
    __syncthreads();
    Frag a = load_frag(R2, 0, 512, kk, lane);
    for (int ni = 0; ni < 4; ++ni) {
      Frag b = load_frag(Panel, w * 64 + ni * 16, 32, 0, lane);
      accY[ni] = wmma_bf16(a, b, accY[ni]);
    }
    __syncthreads();
  }

  // write y reordered to (b, m*h+hh, c) as bf16; 8 consecutive ci -> b128 store
  const int bb = bh >> 3, hh = bh & 7;
  for (int ni = 0; ni < 4; ++ni) {
    int m = w * 64 + ni * 16 + ln;
    Frag t;
    for (int i = 0; i < 8; ++i) t.s[i] = f2bf(accY[ni][i]);
    size_t idx = ((size_t)((bb * LH + m) * NHEAD + hh)) * CH + ci0 + hi * 8;
    *(uv4*)&yR[idx] = t.q[0];
  }
}

// ---------------------------------------------------------------------------
// Kernel 3: out = yR @ w_out^T + b_out   (M=32768, N=512, K=512), f32 out
// ---------------------------------------------------------------------------
__global__ __launch_bounds__(256, 2) void out_gemm(const u16* __restrict__ yR,
                                                   const u16* __restrict__ wo,
                                                   const float* __restrict__ bout,
                                                   float* __restrict__ out) {
  __shared__ __align__(16) u16 As[128 * 32];
  __shared__ __align__(16) u16 Bs[128 * 32];
  const int tid = threadIdx.x, lane = tid & 31, wave = tid >> 5;
  const int wm = wave & 3, wn = wave >> 2;
  const int row0g = blockIdx.x * 128;
  const int col0g = blockIdx.y * 128;

  v8f acc[2][4] = {};
  for (int kk = 0; kk < CH; kk += 32) {
    {
      uv4 ta[2], tb[2];
#pragma unroll
      for (int it = 0; it < 2; ++it) {
        int i = tid + it * 256;
        int r = i >> 2, c = i & 3;
        ta[it] = *(const uv4*)&yR[(size_t)(row0g + r) * CH + kk + c * 8];
        tb[it] = *(const uv4*)&wo[(size_t)(col0g + r) * CH + kk + c * 8];
      }
#pragma unroll
      for (int it = 0; it < 2; ++it) {
        int i = tid + it * 256;
        int r = i >> 2, c = i & 3;
        *(uv4*)&As[r * 32 + c * 8] = ta[it];
        *(uv4*)&Bs[r * 32 + c * 8] = tb[it];
      }
    }
    __syncthreads();
    Frag a0 = load_frag(As, wm * 32, 32, 0, lane);
    Frag a1 = load_frag(As, wm * 32 + 16, 32, 0, lane);
    for (int ni = 0; ni < 4; ++ni) {
      Frag b = load_frag(Bs, wn * 64 + ni * 16, 32, 0, lane);
      acc[0][ni] = wmma_bf16(a0, b, acc[0][ni]);
      acc[1][ni] = wmma_bf16(a1, b, acc[1][ni]);
    }
    __syncthreads();
  }

  // base + i*CH strided stores per tile column: minimal address pressure
  const int hi = lane >> 4, ln = lane & 15;
  for (int mi = 0; mi < 2; ++mi) {
    int gr0 = row0g + wm * 32 + mi * 16 + 8 * hi;
    for (int ni = 0; ni < 4; ++ni) {
      int gc = col0g + wn * 64 + ni * 16 + ln;
      float bias = bout[gc];
      float* dst = out + (size_t)gr0 * CH + gc;
#pragma unroll
      for (int i = 0; i < 8; ++i)
        dst[(size_t)i * CH] = acc[mi][ni][i] + bias;
    }
  }
}

// ---------------------------------------------------------------------------
extern "C" void kernel_launch(void* const* d_in, const int* in_sizes, int n_in,
                              void* d_out, int out_size, void* d_ws, size_t ws_size,
                              hipStream_t stream) {
  const float* x    = (const float*)d_in[0];
  const float* wqkv = (const float*)d_in[1];
  const float* wout = (const float*)d_in[2];
  const float* bout = (const float*)d_in[3];
  float* out = (float*)d_out;
  u16*   ws  = (u16*)d_ws;

  cvt_weights<<<dim3((NQKV * CH + CH * CH) / 256), 256, 0, stream>>>(wqkv, wout, ws);

  qkv_gemm<<<dim3(BATCH * SEQ / 128, NQKV / 128), 256, 0, stream>>>(
      x, ws + WS_WQ, ws + WS_QT, ws + WS_KT, ws + WS_VS);

  attn_fused<<<dim3(CH / 16, BATCH * NHEAD), 256, 0, stream>>>(
      ws + WS_QT, ws + WS_KT, ws + WS_VS, ws + WS_YR);

  out_gemm<<<dim3(BATCH * SEQ / 128, CH / 128), 256, 0, stream>>>(
      ws + WS_YR, ws + WS_WO, bout, out);
}
